// InterFormerEncoder_62826781606557
// MI455X (gfx1250) — matrix operand order repaired and checked by reading, not compiled, not run on hardware
//
#include <hip/hip_runtime.h>
#include <hip/hip_bf16.h>
#include <stdint.h>

#define HH    8
#define HID   256
#define FFD   1024
#define LL    2
#define OFFC  512
#define BB    4
#define CC    200
#define WW    100
#define PE_ROWS 1025
#define MREL  (BB*CC*WW)        /* 80000 */
#define NX    (BB*CC)           /* 800 rows of x */
#define NB    4                 /* N-tiles per wave (16x64 output block) */

typedef __bf16 bf16t;
typedef __attribute__((ext_vector_type(8)))  bf16t v8bf;
typedef __attribute__((ext_vector_type(16))) bf16t v16bf;
typedef __attribute__((ext_vector_type(8)))  float v8f;

// Build the 16x32 bf16 A-fragment for one lane: elements K {hi*16..hi*16+15}
// of the 32-wide K chunk (hi = lane>>4), matching the contiguous B fragment.
static __device__ __forceinline__ v16bf make_afrag(const bf16t* p) {
  v8bf lo = *(const v8bf*)(p);
  v8bf hi = *(const v8bf*)(p + 8);
  return __builtin_shufflevector(lo, hi, 0,1,2,3,4,5,6,7,8,9,10,11,12,13,14,15);
}

// ---------------------------------------------------------------------------
// Generic batched WMMA GEMM:  D[z,m,n] = act( sum_k A[z,m,k]*B[z,n,k] + bias[n] )
// A row-major MxK (bf16), B row-major NxK (bf16, i.e. weight (out,in) layout).
// One wave per 16x(16*NB) output block: the A fragment is loaded once per
// 32-wide K chunk and reused by NB WMMAs (load-issue density 2.5 loads/WMMA).
// Batch z = blockIdx.z decomposed as z1 = z/zdiv, z0 = z%zdiv.
// ---------------------------------------------------------------------------
__global__ void gemm_wmma(const bf16t* __restrict__ A, const bf16t* __restrict__ Bm,
                          const float* __restrict__ bias, void* __restrict__ out,
                          int out_bf16, int relu,
                          int M, int N, int K, int lda, int ldb, int ldc,
                          int zdiv, long sA1, long sA0, long sB1, long sB0,
                          long sC1, long sC0)
{
  const int lane = threadIdx.x;
  const int z  = blockIdx.z;
  const int z1 = z / zdiv, z0 = z % zdiv;
  const bf16t* Ab = A  + z1 * sA1 + z0 * sA0;
  const bf16t* Bb = Bm + z1 * sB1 + z0 * sB0;
  const int m0  = blockIdx.y * 16;
  const int nb0 = blockIdx.x * (16 * NB);
  int arow = m0 + (lane & 15); if (arow >= M) arow = M - 1;   // clamp: dup rows, never stored
  const int hi = lane >> 4;
  const bf16t* Ap = Ab + (long)arow * lda + hi * 16;
  const bf16t* Bp[NB];
#pragma unroll
  for (int j = 0; j < NB; ++j) {
    int bcol = nb0 + j * 16 + (lane & 15); if (bcol >= N) bcol = N - 1;
    Bp[j] = Bb + (long)bcol * ldb + hi * 16;
  }
  v8f acc[NB] = {};
  for (int k0 = 0; k0 < K; k0 += 32) {
    v16bf a = make_afrag(Ap + k0);
#pragma unroll
    for (int j = 0; j < NB; ++j) {
      v16bf b = *(const v16bf*)(Bp[j] + k0);
      acc[j] = __builtin_amdgcn_wmma_f32_16x16x32_bf16(false, a, false, b,
                                                       (short)0, acc[j], false, false);
    }
  }
  const long cbase = z1 * sC1 + z0 * sC0;
#pragma unroll
  for (int j = 0; j < NB; ++j) {
    const int n = nb0 + j * 16 + (lane & 15);
    if (n >= N) continue;
    const float bv = bias ? bias[n] : 0.0f;
#pragma unroll
    for (int r = 0; r < 8; ++r) {
      const int m = m0 + hi * 8 + r;
      if (m < M) {
        float val = acc[j][r] + bv;
        if (relu) val = fmaxf(val, 0.0f);
        const long off = cbase + (long)m * ldc + n;
        if (out_bf16) ((bf16t*)out)[off] = (bf16t)val;
        else          ((float*)out)[off]  = val;
      }
    }
  }
}

// ---------------------------------------------------------------------------
// rel = relu(concat(pe[dss],pe[dse],pe[des],pe[dee]) @ W_fuse^T + b_fuse)
// Gather fused into the GEMM: A row m=(b,c,w) is 4 pe rows of 256 (pe stays
// L2-hot; the 320MB "four" tensor is never materialized). One wave per
// 16x64 output block (NB=4 N-tiles), grid: x=4, y=5000.
// ---------------------------------------------------------------------------
__global__ void rel_gemm_kernel(const bf16t* __restrict__ peb, const bf16t* __restrict__ wfb,
                                const float* __restrict__ bfuse,
                                const int* __restrict__ pos_s, const int* __restrict__ pos_e,
                                const int* __restrict__ lex_s, const int* __restrict__ lex_e,
                                bf16t* __restrict__ rel)
{
  const int lane = threadIdx.x;
  const int m0  = blockIdx.y * 16;
  const int nb0 = blockIdx.x * (16 * NB);
  const int hi = lane >> 4;
  const int m = m0 + (lane & 15);
  const int b   = m / (CC * WW);
  const int rem = m % (CC * WW);
  const int c   = rem / WW;
  const int w   = rem % WW;
  const int ps = pos_s[b * CC + c];
  const int pv = pos_e[b * CC + c];
  const int ls = lex_s[b * WW + w];
  const int le = lex_e[b * WW + w];
  int idx[4];
  idx[0] = ps - ls + OFFC; idx[1] = ps - le + OFFC;
  idx[2] = pv - ls + OFFC; idx[3] = pv - le + OFFC;
#pragma unroll
  for (int s = 0; s < 4; ++s) idx[s] = min(max(idx[s], 0), PE_ROWS - 1);
  v8f acc[NB] = {};
#pragma unroll
  for (int s = 0; s < 4; ++s) {
    const bf16t* Ap = peb + (long)idx[s] * HID + hi * 16;
    for (int kk = 0; kk < HID; kk += 32) {
      v16bf a = make_afrag(Ap + kk);
#pragma unroll
      for (int j = 0; j < NB; ++j) {
        const int ncol = nb0 + j * 16 + (lane & 15);
        v16bf bb = *(const v16bf*)(wfb + (long)ncol * (4 * HID) + s * HID + hi * 16 + kk);
        acc[j] = __builtin_amdgcn_wmma_f32_16x16x32_bf16(false, a, false, bb,
                                                         (short)0, acc[j], false, false);
      }
    }
  }
#pragma unroll
  for (int j = 0; j < NB; ++j) {
    const int ncol = nb0 + j * 16 + (lane & 15);
    const float bv = bfuse[ncol];
#pragma unroll
    for (int r = 0; r < 8; ++r) {
      const int mm = m0 + hi * 8 + r;
      rel[(long)mm * HID + ncol] = (bf16t)fmaxf(acc[j][r] + bv, 0.0f);
    }
  }
}

// ---------------------------- small helper kernels -------------------------
__global__ void cvt_bf16_kernel(bf16t* __restrict__ dst, const float* __restrict__ src, long n) {
  long i = (long)blockIdx.x * blockDim.x + threadIdx.x;
  if (i < n) dst[i] = (bf16t)src[i];
}
__global__ void copy_f32_kernel(float* __restrict__ dst, const float* __restrict__ src, long n) {
  long i = (long)blockIdx.x * blockDim.x + threadIdx.x;
  if (i < n) dst[i] = src[i];
}
// WqT[h][e][d] = Wq[h][d][e]  (bf16), for t = qv @ Wq (contract over d)
__global__ void transpose_wq_kernel(bf16t* __restrict__ dst, const float* __restrict__ src) {
  long i = (long)blockIdx.x * blockDim.x + threadIdx.x;
  if (i < (long)HH * HID * HID) {
    int h = (int)(i >> 16); int e = (int)((i >> 8) & 255); int d = (int)(i & 255);
    dst[i] = (bf16t)src[((long)h << 16) + ((long)d << 8) + e];
  }
}
// vT[b][d][w(pad128)] = v_w[b][w][d], zero pad for K alignment
__global__ void build_vt_kernel(bf16t* __restrict__ dst, const bf16t* __restrict__ vw) {
  long i = (long)blockIdx.x * blockDim.x + threadIdx.x;
  if (i < (long)BB * HID * 128) {
    int b = (int)(i / (HID * 128)); int r = (int)(i % (HID * 128));
    int d = r >> 7; int w = r & 127;
    dst[i] = (w < WW) ? vw[((long)b * WW + w) * HID + d] : (bf16t)0.0f;
  }
}
// qu/qv = bf16(q + u[d]) / bf16(q + v[d]) over (B,C,(h,d)) layout
__global__ void quv_kernel(const float* __restrict__ q, const float* __restrict__ u,
                           const float* __restrict__ v, bf16t* __restrict__ qub,
                           bf16t* __restrict__ qvb) {
  long i = (long)blockIdx.x * blockDim.x + threadIdx.x;
  if (i < (long)NX * HH * HID) {
    int d = (int)(i & (HID - 1));
    float qq = q[i];
    qub[i] = (bf16t)(qq + u[d]);
    qvb[i] = (bf16t)(qq + v[d]);
  }
}
// qvbq[b,c,h] = sum_d (q[b,c,h,d]+v[d]) * bq[h,d]   (8 waves per (b,c) row)
__global__ void qvbq_kernel(const float* __restrict__ q, const float* __restrict__ v,
                            const float* __restrict__ bq, float* __restrict__ out) {
  const int rowbc = blockIdx.x;
  const int h = threadIdx.x >> 5;
  const int lane = threadIdx.x & 31;
  const float* qr  = q  + (long)rowbc * (HH * HID) + h * HID;
  const float* bqh = bq + h * HID;
  float sum = 0.0f;
  for (int d = lane; d < HID; d += 32) sum += (qr[d] + v[d]) * bqh[d];
#pragma unroll
  for (int o = 16; o; o >>= 1) sum += __shfl_xor(sum, o, 32);
  if (lane == 0) out[rowbc * HH + h] = sum;
}
// masked softmax over W per (b,h,c); att zero-padded to 128 for the out-GEMM
__global__ void softmax_kernel(const float* __restrict__ acf, const float* __restrict__ bdf,
                               const float* __restrict__ qvbq,
                               const int* __restrict__ seq_len, const int* __restrict__ lex_num,
                               bf16t* __restrict__ attb) {
  __shared__ float s[128];
  const int row = blockIdx.x;              // (b*H + h)*C + c
  const int c  = row % CC;
  const int bh = row / CC;
  const int h  = bh % HH;
  const int b  = bh / HH;
  const int t  = threadIdx.x;
  const bool vc = c < seq_len[b];
  const int  ln = lex_num[b];
  float sc = -3.0e38f;
  bool valid = false;
  float e = 0.0f;
  if (t < WW) {
    valid = vc && (t < ln);
    float sv = acf[(long)row * WW + t]
             + bdf[((long)(b * CC + c) * HH + h) * WW + t]
             + qvbq[(b * CC + c) * HH + h];
    sc = valid ? sv : -1.0e15f;
  }
  s[t] = sc; __syncthreads();
  for (int o = 64; o; o >>= 1) { if (t < o) s[t] = fmaxf(s[t], s[t + o]); __syncthreads(); }
  const float mx = s[0]; __syncthreads();
  if (t < WW) e = __expf(sc - mx);
  s[t] = e; __syncthreads();
  for (int o = 64; o; o >>= 1) { if (t < o) s[t] += s[t + o]; __syncthreads(); }
  const float inv = 1.0f / s[0];
  const float p = valid ? e * inv : 0.0f;
  attb[(long)row * 128 + t] = (bf16t)p;
}
// x = layernorm(x + add) * g + be; also refresh x in bf16
__global__ void add_ln_kernel(float* __restrict__ x, const float* __restrict__ add,
                              const float* __restrict__ g, const float* __restrict__ be,
                              bf16t* __restrict__ xb) {
  __shared__ float s[HID];
  const long row = blockIdx.x;
  const int t = threadIdx.x;
  float v = x[row * HID + t] + add[row * HID + t];
  s[t] = v; __syncthreads();
  for (int o = 128; o; o >>= 1) { if (t < o) s[t] += s[t + o]; __syncthreads(); }
  const float mean = s[0] * (1.0f / HID); __syncthreads();
  const float d = v - mean;
  s[t] = d * d; __syncthreads();
  for (int o = 128; o; o >>= 1) { if (t < o) s[t] += s[t + o]; __syncthreads(); }
  const float var = s[0] * (1.0f / HID);
  const float y = d * rsqrtf(var + 1e-5f) * g[t] + be[t];
  x[row * HID + t]  = y;
  xb[row * HID + t] = (bf16t)y;
}
__global__ void zero_kernel(float* p) { p[0] = 0.0f; }
// BCE loss over mean-over-heads of A_C (all B*C*W positions, as in reference)
__global__ void loss_kernel(const float* __restrict__ acf, const int* __restrict__ label,
                            float* __restrict__ accum) {
  __shared__ float s[256];
  float local = 0.0f;
  for (long i = (long)blockIdx.x * blockDim.x + threadIdx.x; i < MREL;
       i += (long)gridDim.x * blockDim.x) {
    int b = (int)(i / (CC * WW));
    int rem = (int)(i % (CC * WW));
    int c = rem / WW;
    int w = rem % WW;
    float msum = 0.0f;
#pragma unroll
    for (int h = 0; h < HH; ++h)
      msum += acf[((long)(b * HH + h) * CC + c) * WW + w];
    const float m = msum * (1.0f / HH);
    const float y = (float)label[i];
    const float lcore = log1pf(__expf(-fabsf(m)));
    const float logp   = fmaxf(-(lcore + fmaxf(-m, 0.0f)), -100.0f);  // -softplus(-m)
    const float log1mp = fmaxf(-(lcore + fmaxf( m, 0.0f)), -100.0f);  // -softplus(m)
    local += y * logp + (1.0f - y) * log1mp;
  }
  s[threadIdx.x] = local; __syncthreads();
  for (int o = 128; o; o >>= 1) { if ((int)threadIdx.x < o) s[threadIdx.x] += s[threadIdx.x + o]; __syncthreads(); }
  if (threadIdx.x == 0) atomicAdd(accum, s[0]);
}
__global__ void finalize_loss_kernel(const float* __restrict__ accum, float* __restrict__ out) {
  out[0] = -accum[0] / (float)MREL;
}

// ---------------------------------------------------------------------------
static inline size_t alignup(size_t x) { return (x + 255) & ~(size_t)255; }

static void launch_gemm(hipStream_t st, const bf16t* A, const bf16t* B, const float* bias,
                        void* out, int obf, int relu, int M, int N, int K,
                        int lda, int ldb, int ldc, int Z, int zdiv,
                        long sA1, long sA0, long sB1, long sB0, long sC1, long sC0) {
  const int ntiles = (N + 15) / 16;
  dim3 grid((ntiles + NB - 1) / NB, (M + 15) / 16, Z);
  gemm_wmma<<<grid, 32, 0, st>>>(A, B, bias, out, obf, relu, M, N, K, lda, ldb, ldc,
                                 zdiv, sA1, sA0, sB1, sB0, sC1, sC0);
}

extern "C" void kernel_launch(void* const* d_in, const int* in_sizes, int n_in,
                              void* d_out, int out_size, void* d_ws, size_t ws_size,
                              hipStream_t stream) {
  (void)in_sizes; (void)n_in; (void)out_size; (void)ws_size;
  // ---- inputs (setup_inputs() dict order; params flattened in insertion order)
  const int*   label  = (const int*)d_in[0];
  const float* chars  = (const float*)d_in[1];
  const float* words  = (const float*)d_in[2];
  const int*   pos_s  = (const int*)d_in[3];
  const int*   pos_e  = (const int*)d_in[4];
  const int*   lex_s  = (const int*)d_in[5];
  const int*   lex_e  = (const int*)d_in[6];
  const int*   seqlen = (const int*)d_in[7];
  const int*   lexnum = (const int*)d_in[8];
  const float* pe     = (const float*)d_in[9];
  const float* Wfuse  = (const float*)d_in[10];
  const float* bfuse  = (const float*)d_in[11];

  // ---- workspace carve
  uint8_t* wsp = (uint8_t*)d_ws;
  auto take = [&](size_t bytes) -> uint8_t* { uint8_t* p = wsp; wsp += alignup(bytes); return p; };
  bf16t* peb    = (bf16t*)take((size_t)PE_ROWS * HID * 2);
  bf16t* wfb    = (bf16t*)take((size_t)HID * 4 * HID * 2);
  bf16t* wordsb = (bf16t*)take((size_t)BB * WW * HID * 2);
  float* xf     = (float*)take((size_t)NX * HID * 4);
  bf16t* xb     = (bf16t*)take((size_t)NX * HID * 2);
  bf16t* rel    = (bf16t*)take((size_t)MREL * HID * 2);
  bf16t* relr   = (bf16t*)take((size_t)MREL * HID * 2);
  bf16t* wqb    = (bf16t*)take((size_t)HH * HID * HID * 2);
  bf16t* wqtb   = (bf16t*)take((size_t)HH * HID * HID * 2);
  bf16t* wkb    = (bf16t*)take((size_t)HID * HID * 2);
  bf16t* wvb    = (bf16t*)take((size_t)HID * HID * 2);
  bf16t* wrb    = (bf16t*)take((size_t)HID * HID * 2);
  bf16t* wpb    = (bf16t*)take((size_t)HID * HH * HID * 2);
  bf16t* w1b    = (bf16t*)take((size_t)FFD * HID * 2);
  bf16t* w2b    = (bf16t*)take((size_t)HID * FFD * 2);
  bf16t* kwb    = (bf16t*)take((size_t)BB * WW * HID * 2);
  bf16t* vwb    = (bf16t*)take((size_t)BB * WW * HID * 2);
  bf16t* vtb    = (bf16t*)take((size_t)BB * HID * 128 * 2);
  float* qf     = (float*)take((size_t)NX * HH * HID * 4);
  bf16t* qub    = (bf16t*)take((size_t)NX * HH * HID * 2);
  bf16t* qvb    = (bf16t*)take((size_t)NX * HH * HID * 2);
  float* qvbqf  = (float*)take((size_t)NX * HH * 4);
  float* acf    = (float*)take((size_t)BB * HH * CC * WW * 4);
  bf16t* tb     = (bf16t*)take((size_t)NX * HH * HID * 2);
  float* bdf    = (float*)take((size_t)NX * HH * WW * 4);
  bf16t* attb   = (bf16t*)take((size_t)BB * HH * CC * 128 * 2);
  bf16t* outb   = (bf16t*)take((size_t)NX * HH * HID * 2);
  float* aof    = (float*)take((size_t)NX * HID * 4);
  bf16t* ff1b   = (bf16t*)take((size_t)NX * FFD * 2);
  float* accum  = (float*)take(4);

  auto cvt  = [&](bf16t* d, const float* s, long n) {
    cvt_bf16_kernel<<<(unsigned)((n + 255) / 256), 256, 0, stream>>>(d, s, n);
  };

  // ---- preamble: bf16 conversions, x init
  cvt(peb, pe, (long)PE_ROWS * HID);
  cvt(wfb, Wfuse, (long)HID * 4 * HID);
  cvt(wordsb, words, (long)BB * WW * HID);
  copy_f32_kernel<<<(NX * HID + 255) / 256, 256, 0, stream>>>(xf, chars, (long)NX * HID);
  cvt(xb, chars, (long)NX * HID);

  // ---- rel GEMM (gather fused), M=80000, N=256, K=1024
  {
    dim3 grid(256 / (16 * NB), MREL / 16, 1);
    rel_gemm_kernel<<<grid, 32, 0, stream>>>(peb, wfb, bfuse, pos_s, pos_e, lex_s, lex_e, rel);
  }

  for (int l = 0; l < LL; ++l) {
    const float* Wq = (const float*)d_in[12] + (size_t)l * HH * HID * HID;
    const float* bq = (const float*)d_in[13] + (size_t)l * HH * HID;
    const float* Wk = (const float*)d_in[14] + (size_t)l * HID * HID;
    const float* bk = (const float*)d_in[15] + (size_t)l * HID;
    const float* Wv = (const float*)d_in[16] + (size_t)l * HID * HID;
    const float* bv = (const float*)d_in[17] + (size_t)l * HID;
    const float* Wr = (const float*)d_in[18] + (size_t)l * HID * HID;
    const float* br = (const float*)d_in[19] + (size_t)l * HID;
    const float* uu = (const float*)d_in[20] + (size_t)l * HID;
    const float* vv = (const float*)d_in[21] + (size_t)l * HID;
    const float* Wp = (const float*)d_in[22] + (size_t)l * HID * HH * HID;
    const float* g1 = (const float*)d_in[23] + (size_t)l * HID;
    const float* be1= (const float*)d_in[24] + (size_t)l * HID;
    const float* g2 = (const float*)d_in[25] + (size_t)l * HID;
    const float* be2= (const float*)d_in[26] + (size_t)l * HID;
    const float* W1 = (const float*)d_in[27] + (size_t)l * FFD * HID;
    const float* b1 = (const float*)d_in[28] + (size_t)l * FFD;
    const float* W2 = (const float*)d_in[29] + (size_t)l * HID * FFD;
    const float* b2 = (const float*)d_in[30] + (size_t)l * HID;

    // layer weight conversions
    cvt(wqb, Wq, (long)HH * HID * HID);
    transpose_wq_kernel<<<(HH * HID * HID + 255) / 256, 256, 0, stream>>>(wqtb, Wq);
    cvt(wkb, Wk, (long)HID * HID);
    cvt(wvb, Wv, (long)HID * HID);
    cvt(wrb, Wr, (long)HID * HID);
    cvt(wpb, Wp, (long)HID * HH * HID);
    cvt(w1b, W1, (long)FFD * HID);
    cvt(w2b, W2, (long)HID * FFD);

    // k_w / v_w: (B*W,256) = words @ W^T + b
    launch_gemm(stream, wordsb, wkb, bk, kwb, 1, 0, BB * WW, HID, HID, HID, HID, HID,
                1, 1, 0, 0, 0, 0, 0, 0);
    launch_gemm(stream, wordsb, wvb, bv, vwb, 1, 0, BB * WW, HID, HID, HID, HID, HID,
                1, 1, 0, 0, 0, 0, 0, 0);
    build_vt_kernel<<<(BB * HID * 128 + 255) / 256, 256, 0, stream>>>(vtb, vwb);

    // rel_r = rel @ Wr^T + br   (M=80000)
    launch_gemm(stream, rel, wrb, br, relr, 1, 0, MREL, HID, HID, HID, HID, HID,
                1, 1, 0, 0, 0, 0, 0, 0);

    // q (f32, layout (b,c,(h,d))): M=800, N=2048, K=256
    launch_gemm(stream, xb, wqb, bq, qf, 0, 0, NX, HH * HID, HID, HID, HID, HH * HID,
                1, 1, 0, 0, 0, 0, 0, 0);
    quv_kernel<<<(NX * HH * HID + 255) / 256, 256, 0, stream>>>(qf, uu, vv, qub, qvb);
    qvbq_kernel<<<NX, HH * 32, 0, stream>>>(qf, vv, bq, qvbqf);

    // A_C[b,h,c,w] = qu . k_w : z=(b,h), M=200, N=100, K=256
    launch_gemm(stream, qub, kwb, nullptr, acf, 0, 0, CC, WW, HID,
                HH * HID, HID, WW, BB * HH, HH,
                (long)CC * HH * HID, HID, (long)WW * HID, 0,
                (long)HH * CC * WW, (long)CC * WW);

    // t[b,c,h,e] = qv @ Wq (contract d): z=h, M=800, N=256, K=256
    launch_gemm(stream, qvb, wqtb, nullptr, tb, 1, 0, NX, HID, HID,
                HH * HID, HID, HH * HID, HH, HH,
                0, HID, 0, (long)HID * HID, 0, HID);

    // B_D[b,c,h,w] = t . rel_r : z=(b,c), M=8, N=100, K=256
    launch_gemm(stream, tb, relr, nullptr, bdf, 0, 0, HH, WW, HID,
                HID, HID, WW, NX, 1,
                (long)HH * HID, 0, (long)WW * HID, 0, (long)HH * WW, 0);

    // masked softmax -> att (bf16, K padded to 128)
    softmax_kernel<<<BB * HH * CC, 128, 0, stream>>>(acf, bdf, qvbqf, seqlen, lexnum, attb);

    // out[b,c,(h,d)] = att @ v_w : z=(b,h), M=200, N=256, K=128
    launch_gemm(stream, attb, vtb, nullptr, outb, 1, 0, CC, HID, 128,
                128, 128, HH * HID, BB * HH, HH,
                (long)HH * CC * 128, (long)CC * 128, (long)HID * 128, 0,
                (long)CC * HH * HID, HID);

    // attn_out = out @ Wproj^T : M=800, N=256, K=2048
    launch_gemm(stream, outb, wpb, nullptr, aof, 0, 0, NX, HID, HH * HID,
                HH * HID, HH * HID, HID, 1, 1, 0, 0, 0, 0, 0, 0);

    add_ln_kernel<<<NX, HID, 0, stream>>>(xf, aof, g1, be1, xb);

    // FF: relu(x@W1^T+b1) @ W2^T + b2
    launch_gemm(stream, xb, w1b, b1, ff1b, 1, 1, NX, FFD, HID, HID, HID, FFD,
                1, 1, 0, 0, 0, 0, 0, 0);
    launch_gemm(stream, ff1b, w2b, b2, aof, 0, 0, NX, HID, FFD, FFD, FFD, HID,
                1, 1, 0, 0, 0, 0, 0, 0);

    add_ln_kernel<<<NX, HID, 0, stream>>>(xf, aof, g2, be2, xb);

    if (l == LL - 1) {  // loss uses last layer's (unmasked) A_C
      zero_kernel<<<1, 1, 0, stream>>>(accum);
      loss_kernel<<<256, 256, 0, stream>>>(acf, label, accum);
      finalize_loss_kernel<<<1, 1, 0, stream>>>(accum, (float*)d_out + (size_t)NX * HID);
    }
  }

  // x -> d_out[0 .. 204799]
  copy_f32_kernel<<<(NX * HID + 255) / 256, 256, 0, stream>>>((float*)d_out, xf, (long)NX * HID);
}
